// CustomGNNLayer_67173288510040
// MI455X (gfx1250) — matrix operand order repaired and checked by reading, not compiled, not run on hardware
//
#include <hip/hip_runtime.h>
#include <hip/hip_bf16.h>
#include <math.h>

typedef _Float16 v16h __attribute__((ext_vector_type(16)));
typedef _Float16 v8h  __attribute__((ext_vector_type(8)));
typedef float    v8f  __attribute__((ext_vector_type(8)));

// Problem dims (compile-time constants from the reference)
static constexpr int Qd = 64, Kd = 2, Nd = 32, Md = 64;
static constexpr int Ed = 256, Dd = 1024, Rd = 200, Sd = 8192;
static constexpr int AS_STRIDE = Ed + 8;   // f16 elems; 528B rows = 33*16B, +4 banks/row

// ---------------------------------------------------------------------------
// Kernel 1: WnT[d][e] = (f16) Wn[e][d]   (transpose+convert, LDS tiled)
// grid 256 (=32 d-tiles * 8 e-tiles), block 256
__global__ void k_wnT(const float* __restrict__ Wn, _Float16* __restrict__ WnT) {
    __shared__ float tile[32][33];
    const int bd = (blockIdx.x & 31) * 32;   // d tile base
    const int be = (blockIdx.x >> 5) * 32;   // e tile base
    const int tx = threadIdx.x & 31, ty = threadIdx.x >> 5;  // ty: 0..7
    #pragma unroll
    for (int j = 0; j < 32; j += 8)
        tile[ty + j][tx] = Wn[(size_t)(be + ty + j) * Dd + bd + tx];
    __syncthreads();
    #pragma unroll
    for (int j = 0; j < 32; j += 8)
        WnT[(size_t)(bd + ty + j) * Ed + be + tx] = (_Float16)tile[tx][ty + j];
}

// ---------------------------------------------------------------------------
// Kernel 2: gq[q][d] = tanh(hs[gnn_idx[q]] @ Wq + bq)   grid 64*4, block 256
__global__ void k_gq(const float* __restrict__ hs, const int* __restrict__ gnn_idx,
                     const float* __restrict__ Wq, const float* __restrict__ bq,
                     float* __restrict__ gq) {
    __shared__ float hsrow[Dd];
    const int q = blockIdx.x >> 2;
    const int d = ((blockIdx.x & 3) << 8) + threadIdx.x;
    const int ri = gnn_idx[q];
    for (int i = threadIdx.x; i < Dd; i += 256) hsrow[i] = hs[(size_t)ri * Dd + i];
    __syncthreads();
    float acc = bq[d];
    #pragma unroll 4
    for (int i = 0; i < Dd; ++i) acc += hsrow[i] * Wq[(size_t)i * Dd + d];
    gq[q * Dd + d] = tanhf(acc);
}

// ---------------------------------------------------------------------------
// Kernel 3: rel_prob[q] = softmax(hs[rel_idx[q]] @ Wc + bc)   grid 64, block 256
__global__ void k_relprob(const float* __restrict__ hs, const int* __restrict__ rel_idx,
                          const float* __restrict__ Wc, const float* __restrict__ bc,
                          float* __restrict__ rel_prob) {
    __shared__ float hsrow[Dd];
    __shared__ float lg[Rd];
    __shared__ float ms[2];
    const int q = blockIdx.x;
    const int ri = rel_idx[q];
    for (int i = threadIdx.x; i < Dd; i += 256) hsrow[i] = hs[(size_t)ri * Dd + i];
    __syncthreads();
    if (threadIdx.x < Rd) {
        float acc = bc[threadIdx.x];
        #pragma unroll 4
        for (int i = 0; i < Dd; ++i) acc += hsrow[i] * Wc[(size_t)i * Rd + threadIdx.x];
        lg[threadIdx.x] = acc;
    }
    __syncthreads();
    if (threadIdx.x == 0) {
        float mx = -INFINITY;
        for (int r = 0; r < Rd; ++r) mx = fmaxf(mx, lg[r]);
        float s = 0.f;
        for (int r = 0; r < Rd; ++r) s += __expf(lg[r] - mx);
        ms[0] = mx; ms[1] = s;
    }
    __syncthreads();
    if (threadIdx.x < Rd)
        rel_prob[q * Rd + threadIdx.x] = __expf(lg[threadIdx.x] - ms[0]) / ms[1];
}

// ---------------------------------------------------------------------------
// Kernel 4 (the big one): per (q,k,n) group, WMMA GEMM [64x256]x[256x1024],
// fused tanh(.+bn) * gq reduction over D  ->  dots[Q,K,N,M]
// grid Q*K*N = 4096, block 256 (8 waves)
__global__ void k_dots(const float* __restrict__ nodes, const _Float16* __restrict__ WnT,
                       const float* __restrict__ bn, const float* __restrict__ gq,
                       float* __restrict__ dots) {
    __shared__ _Float16 As[Md * AS_STRIDE];
    __shared__ float red[Md][64];
    const int tid = threadIdx.x;
    const int g = blockIdx.x;                 // (q*K + k)*N + n
    const int q = g / (Kd * Nd);
    const size_t nbase = (size_t)g * Md * Ed;

    // Stage A tile (64x256 f32 -> f16, padded stride) + zero the reduce buffer
    for (int i = tid; i < Md * Ed; i += 256) {
        const int r = i >> 8, c = i & 255;
        As[r * AS_STRIDE + c] = (_Float16)nodes[nbase + i];
    }
    for (int i = tid; i < Md * 64; i += 256) ((float*)red)[i] = 0.f;
    __syncthreads();

    const int wave  = tid >> 5;
    const int lane  = tid & 31;
    const int half  = lane >> 4;    // K-half select
    const int l15   = lane & 15;
    const int rtile = wave & 3;     // 4 row tiles of 16
    const int par   = wave >> 2;    // d-block parity
    const int arow  = rtile * 16 + l15;
    const int koffA = half ? 16 : 0;
    const int koffB = half ? 16 : 0;
    const float* gq_row = gq + q * Dd;

    float partial[8];
    #pragma unroll
    for (int v = 0; v < 8; ++v) partial[v] = 0.f;

    for (int db = par; db < Dd / 16; db += 2) {
        const int d = db * 16 + l15;          // this lane's output column
        const _Float16* bptr = WnT + (size_t)d * Ed;
        v8f c = {};
        #pragma unroll
        for (int ks = 0; ks < 8; ++ks) {
            const int kb = ks * 32;
            v8h alo = *(const v8h*)(&As[arow * AS_STRIDE + kb + koffA]);
            v8h ahi = *(const v8h*)(&As[arow * AS_STRIDE + kb + koffA + 8]);
            v8h blo = *(const v8h*)(&bptr[kb + koffB]);
            v8h bhi = *(const v8h*)(&bptr[kb + koffB + 8]);
            v16h a = __builtin_shufflevector(alo, ahi, 0,1,2,3,4,5,6,7,8,9,10,11,12,13,14,15);
            v16h b = __builtin_shufflevector(blo, bhi, 0,1,2,3,4,5,6,7,8,9,10,11,12,13,14,15);
            c = __builtin_amdgcn_wmma_f32_16x16x32_f16(false, a, false, b,
                                                       (short)0, c, false, false);
        }
        const float bnv = bn[d];
        const float gqv = gq_row[d];
        #pragma unroll
        for (int v = 0; v < 8; ++v)
            partial[v] += tanhf(c[v] + bnv) * gqv;
    }

    // Scatter per-lane partials: row = rtile*16 + (half?8:0) + v, unique column
    {
        const int rbase = rtile * 16 + (half ? 8 : 0);
        const int col = par * 32 + lane;
        #pragma unroll
        for (int v = 0; v < 8; ++v) red[rbase + v][col] = partial[v];
    }
    __syncthreads();
    if (tid < Md) {
        float s = 0.f;
        #pragma unroll 8
        for (int c2 = 0; c2 < 64; ++c2) s += red[tid][c2];
        dots[(size_t)g * Md + tid] = s;
    }
}

// ---------------------------------------------------------------------------
// Kernel 5: per (q,k): group softmax -> *prob/0.1 -> global softmax -> mask
//           -> masked weighted mean over nodes  => mean_emb[Q,K,E]
// grid Q*K = 128, block 256
__global__ void k_attn_pool(const float* __restrict__ nodes, const float* __restrict__ dots,
                            const float* __restrict__ rel_prob, const int* __restrict__ prob_idx,
                            float* __restrict__ mean_emb) {
    __shared__ float sd[Nd * Md];     // 2048: dots -> attn -> logits -> wa
    __shared__ float mf[Nd * Md];     // mask (0/1)
    __shared__ float rbuf[256];
    const int tid = threadIdx.x;
    const int qk = blockIdx.x;                       // q*K + k
    const int q = qk / Kd;
    const size_t dbase = (size_t)qk * Nd * Md;
    const size_t nbase = (size_t)qk * Nd * Md * Ed;

    for (int i = tid; i < Nd * Md; i += 256) {
        sd[i] = dots[dbase + i];
        mf[i] = (nodes[nbase + (size_t)i * Ed] != 0.f) ? 1.f : 0.f;
    }
    __syncthreads();

    // per-group softmax over M, then scale by probs/0.1
    if (tid < Nd) {
        const int n = tid;
        float mx = -INFINITY;
        for (int m = 0; m < Md; ++m) mx = fmaxf(mx, sd[n * Md + m]);
        float s = 0.f;
        for (int m = 0; m < Md; ++m) s += __expf(sd[n * Md + m] - mx);
        const float p = rel_prob[q * Rd + prob_idx[qk * Nd + n]];
        const float scale = p * 10.f / s;            // /0.1 temperature
        for (int m = 0; m < Md; ++m)
            sd[n * Md + m] = __expf(sd[n * Md + m] - mx) * scale;
    }
    __syncthreads();

    // global softmax over all N*M slots
    float lm = -INFINITY;
    for (int i = tid; i < Nd * Md; i += 256) lm = fmaxf(lm, sd[i]);
    rbuf[tid] = lm; __syncthreads();
    for (int s = 128; s > 0; s >>= 1) {
        if (tid < s) rbuf[tid] = fmaxf(rbuf[tid], rbuf[tid + s]);
        __syncthreads();
    }
    const float gmax = rbuf[0];
    __syncthreads();
    float ls = 0.f;
    for (int i = tid; i < Nd * Md; i += 256) ls += __expf(sd[i] - gmax);
    rbuf[tid] = ls; __syncthreads();
    for (int s = 128; s > 0; s >>= 1) {
        if (tid < s) rbuf[tid] += rbuf[tid + s];
        __syncthreads();
    }
    const float inv = 1.f / rbuf[0];
    __syncthreads();
    for (int i = tid; i < Nd * Md; i += 256)
        sd[i] = __expf(sd[i] - gmax) * inv * mf[i];  // masked weights
    __syncthreads();

    // weighted mean over the 2048 slots; one feature per thread (E=256)
    const int e = tid;
    float acc = 0.f;
    #pragma unroll 4
    for (int slot = 0; slot < Nd * Md; ++slot)
        acc += nodes[nbase + (size_t)slot * Ed + e] * sd[slot];
    mean_emb[(size_t)qk * Ed + e] = acc * (1.f / (Nd * Md));
}

// ---------------------------------------------------------------------------
// Kernel 6: copy hidden_states -> d_out (float4)   grid 8192, block 256
__global__ void k_copy(const float4* __restrict__ src, float4* __restrict__ dst) {
    const size_t i = (size_t)blockIdx.x * 256 + threadIdx.x;
    dst[i] = src[i];
}

// Kernel 7: d_out[gnn_idx[q]] += tanh(pooled @ Wg + bg)   grid 64*4, block 256
__global__ void k_out(const float* __restrict__ mean_emb, const int* __restrict__ gnn_idx,
                      const float* __restrict__ Wg, const float* __restrict__ bg,
                      float* __restrict__ out) {
    __shared__ float pe[Ed];
    const int q = blockIdx.x >> 2;
    const int d = ((blockIdx.x & 3) << 8) + threadIdx.x;
    pe[threadIdx.x] = 0.5f * (mean_emb[(size_t)(q * Kd + 0) * Ed + threadIdx.x] +
                              mean_emb[(size_t)(q * Kd + 1) * Ed + threadIdx.x]);
    __syncthreads();
    float acc = bg[d];
    #pragma unroll 4
    for (int e = 0; e < Ed; ++e) acc += pe[e] * Wg[(size_t)e * Dd + d];
    const int row = gnn_idx[q];
    out[(size_t)row * Dd + d] += tanhf(acc);
}

// ---------------------------------------------------------------------------
extern "C" void kernel_launch(void* const* d_in, const int* in_sizes, int n_in,
                              void* d_out, int out_size, void* d_ws, size_t ws_size,
                              hipStream_t stream) {
    const float* hs       = (const float*)d_in[0];
    const float* nodes    = (const float*)d_in[1];
    const int*   prob_idx = (const int*)  d_in[2];
    const int*   gnn_idx  = (const int*)  d_in[3];
    const int*   rel_idx  = (const int*)  d_in[4];
    const float* Wc = (const float*)d_in[5];
    const float* bc = (const float*)d_in[6];
    const float* Wq = (const float*)d_in[7];
    const float* bq = (const float*)d_in[8];
    const float* Wn = (const float*)d_in[9];
    const float* bn = (const float*)d_in[10];
    const float* Wg = (const float*)d_in[11];
    const float* bg = (const float*)d_in[12];
    float* out = (float*)d_out;

    char* w = (char*)d_ws;
    _Float16* ws_WnT  = (_Float16*)(w);                       // 1024*256*2 = 512 KiB
    float* ws_gq      = (float*)(w + (512 << 10));            // 64*1024*4 = 256 KiB
    float* ws_relprob = (float*)(w + (768 << 10));            // 64*200*4  (pad to 64 KiB)
    float* ws_dots    = (float*)(w + (832 << 10));            // 262144*4 = 1 MiB
    float* ws_meanemb = (float*)(w + (1856 << 10));           // 128*256*4 = 128 KiB

    k_wnT   <<<256,  256, 0, stream>>>(Wn, ws_WnT);
    k_copy  <<<8192, 256, 0, stream>>>((const float4*)hs, (float4*)out);
    k_gq    <<<Qd*4, 256, 0, stream>>>(hs, gnn_idx, Wq, bq, ws_gq);
    k_relprob<<<Qd,  256, 0, stream>>>(hs, rel_idx, Wc, bc, ws_relprob);
    k_dots  <<<Qd*Kd*Nd, 256, 0, stream>>>(nodes, ws_WnT, bn, ws_gq, ws_dots);
    k_attn_pool<<<Qd*Kd, 256, 0, stream>>>(nodes, ws_dots, ws_relprob, prob_idx, ws_meanemb);
    k_out   <<<Qd*4, 256, 0, stream>>>(ws_meanemb, gnn_idx, Wg, bg, out);
    (void)in_sizes; (void)n_in; (void)out_size; (void)ws_size;
}
